// MambaBlock_22823456211745
// MI455X (gfx1250) — matrix operand order repaired and checked by reading, not compiled, not run on hardware
//
#include <hip/hip_runtime.h>

// ---------------- types for WMMA ----------------
typedef __attribute__((ext_vector_type(16))) __bf16 v16bf;
typedef __attribute__((ext_vector_type(8)))  __bf16 v8bf;
typedef __attribute__((ext_vector_type(8)))  float  v8f;

// ---------------- problem constants ----------------
#define B_  2
#define L_  2048
#define DM  768
#define DS  32
#define DC  7
#define DI  1536
#define BL  (B_ * L_)       // 4096 rows
#define NXZ (2 * DI)        // 3072
#define NP  (2 * DS + DI)   // 1600

// GEMM block tiling
#define BM 128
#define BN 64
#define BK 32

// ============================================================
// CDNA5 async global->LDS copy helpers (ASYNCcnt-tracked)
// ============================================================
__device__ __forceinline__ void async_copy_b128(unsigned lds_addr, const void* gaddr)
{
    asm volatile("global_load_async_to_lds_b128 %0, %1, off"
                 :
                 : "v"(lds_addr), "v"((unsigned long long)(size_t)gaddr)
                 : "memory");
}
__device__ __forceinline__ void wait_async_zero()
{
    asm volatile("s_wait_asynccnt 0x0" ::: "memory");
}

// ============================================================
// f32 -> bf16 convert (weights)
// ============================================================
__global__ __launch_bounds__(256) void cvt_f32_bf16_kernel(
    const float* __restrict__ in, __bf16* __restrict__ out, int n)
{
    int i = blockIdx.x * 256 + threadIdx.x;
    if (i < n) out[i] = (__bf16)in[i];
}

// ============================================================
// LayerNorm over DM=768, output bf16
// one block (256 thr, 8 waves) per row
// ============================================================
__global__ __launch_bounds__(256) void layernorm_bf16_kernel(
    const float* __restrict__ x, const float* __restrict__ g,
    const float* __restrict__ bta, __bf16* __restrict__ out)
{
    const int row  = blockIdx.x;
    const int lane = threadIdx.x & 31;
    const int wave = threadIdx.x >> 5;
    const float* xr = x + (size_t)row * DM;

    float s = 0.f, s2 = 0.f;
    for (int i = threadIdx.x; i < DM; i += 256) {
        float v = xr[i]; s += v; s2 += v * v;
    }
    #pragma unroll
    for (int off = 16; off; off >>= 1) {
        s  += __shfl_xor(s,  off, 32);
        s2 += __shfl_xor(s2, off, 32);
    }
    __shared__ float ls[8], ls2[8];
    if (lane == 0) { ls[wave] = s; ls2[wave] = s2; }
    __syncthreads();
    if (wave == 0) {
        s  = (lane < 8) ? ls[lane]  : 0.f;
        s2 = (lane < 8) ? ls2[lane] : 0.f;
        #pragma unroll
        for (int off = 4; off; off >>= 1) {
            s  += __shfl_xor(s,  off, 32);
            s2 += __shfl_xor(s2, off, 32);
        }
        if (lane == 0) { ls[0] = s; ls2[0] = s2; }
    }
    __syncthreads();
    const float mean = ls[0] * (1.f / DM);
    const float var  = ls2[0] * (1.f / DM) - mean * mean;
    const float rstd = rsqrtf(var + 1e-5f);

    __bf16* orow = out + (size_t)row * DM;
    for (int i = threadIdx.x; i < DM; i += 256) {
        float v = (xr[i] - mean) * rstd * g[i] + bta[i];
        orow[i] = (__bf16)v;
    }
}

// ============================================================
// bf16 WMMA GEMM:  C[M,N] = A[M,K] @ Bw[N,K]^T  (+ optional residual)
//
// Block 256 thr (8 waves), block tile 128x64, K-step 32.
// Wave grid 4(M) x 2(N); each wave owns a 32x32 tile = 2x2 WMMA accs.
// A/B K-tiles staged in LDS via global_load_async_to_lds_b128,
// double-buffered (copy of tile k+1 overlaps WMMA of tile k).
// Requires M%128==0, N%64==0, K%32==0 (true for all three GEMMs).
// ============================================================
__global__ __launch_bounds__(256) void gemm_bf16_wmma_kernel(
    const __bf16* __restrict__ A, const __bf16* __restrict__ Bw,
    float* __restrict__ C, const float* __restrict__ residual,
    int M, int N, int K)
{
    __shared__ alignas(128) __bf16 sA[2][BM][BK];   // 16 KB
    __shared__ alignas(128) __bf16 sB[2][BN][BK];   //  8 KB

    const int nbn  = N / BN;
    const int bm   = blockIdx.x / nbn;
    const int bn   = blockIdx.x % nbn;
    const int row0 = bm * BM;
    const int col0 = bn * BN;

    const int lane = threadIdx.x & 31;
    const int wave = threadIdx.x >> 5;
    const int wm   = wave >> 1;   // 0..3
    const int wn   = wave & 1;    // 0..1
    const int half = lane >> 4;   // 0/1 lane group
    const int l15  = lane & 15;

    // cooperative copy of one K-tile into LDS buffer `buf`
    auto copy_tile = [&](int buf, int k0) {
        {   // A: 128x32 bf16 = 8KB -> 2 x b128 per thread
            const int r  = threadIdx.x >> 1;
            const int kh = (threadIdx.x & 1) * 16;
            const __bf16* src = A + (size_t)(row0 + r) * K + k0 + kh;
            async_copy_b128((unsigned)(size_t)&sA[buf][r][kh],     src);
            async_copy_b128((unsigned)(size_t)&sA[buf][r][kh + 8], src + 8);
        }
        {   // B (rows of Bw): 64x32 bf16 = 4KB -> 1 x b128 per thread
            const int c  = threadIdx.x >> 2;
            const int kh = (threadIdx.x & 3) * 8;
            const __bf16* src = Bw + (size_t)(col0 + c) * K + k0 + kh;
            async_copy_b128((unsigned)(size_t)&sB[buf][c][kh], src);
        }
    };

    v8f acc[2][2] = {};
    const int nk = K / BK;

    copy_tile(0, 0);
    for (int kt = 0; kt < nk; ++kt) {
        wait_async_zero();      // own async copies done
        __syncthreads();        // everyone's copies visible

        const int buf = kt & 1;
        if (kt + 1 < nk) copy_tile(buf ^ 1, (kt + 1) * BK);

        // A fragments (contiguous K, matching B):
        //   a[0..15]=K(16h..16h+15), lane l15 = row
        v16bf afrag[2];
        #pragma unroll
        for (int m = 0; m < 2; ++m) {
            const int row = wm * 32 + m * 16 + l15;
            v8bf lo = *(const v8bf*)&sA[buf][row][16 * half];
            v8bf hi = *(const v8bf*)&sA[buf][row][16 * half + 8];
            #pragma unroll
            for (int i = 0; i < 8; ++i) { afrag[m][i] = lo[i]; afrag[m][8 + i] = hi[i]; }
        }
        // B fragments: lane l15 = output column, b[0..15]=K(16h..16h+15)
        v16bf bfrag[2];
        #pragma unroll
        for (int n = 0; n < 2; ++n) {
            const int col = wn * 32 + n * 16 + l15;
            bfrag[n] = *(const v16bf*)&sB[buf][col][16 * half];
        }

        #pragma unroll
        for (int m = 0; m < 2; ++m)
            #pragma unroll
            for (int n = 0; n < 2; ++n)
                acc[m][n] = __builtin_amdgcn_wmma_f32_16x16x32_bf16(
                                false, afrag[m], false, bfrag[n],
                                (short)0, acc[m][n], false, false);
    }

    // D layout: VGPR v, lanes 0-15 -> M=+v, lanes 16-31 -> M=+v+8
    #pragma unroll
    for (int m = 0; m < 2; ++m) {
        #pragma unroll
        for (int n = 0; n < 2; ++n) {
            const int colg = col0 + wn * 32 + n * 16 + l15;
            #pragma unroll
            for (int v = 0; v < 8; ++v) {
                const int rowg = row0 + wm * 32 + m * 16 + v + 8 * half;
                const size_t idx = (size_t)rowg * N + colg;
                float val = acc[m][n][v];
                if (residual) val += residual[idx];
                C[idx] = val;
            }
        }
    }
}

// ============================================================
// depthwise causal conv (DC=7) + bias + SiLU over xc part of xz
// ============================================================
__global__ __launch_bounds__(256) void conv_silu_kernel(
    const float* __restrict__ xz, const float* __restrict__ cw,
    const float* __restrict__ cb,
    float* __restrict__ xc_f, __bf16* __restrict__ xc_b)
{
    const size_t gid = (size_t)blockIdx.x * 256 + threadIdx.x;
    if (gid >= (size_t)BL * DI) return;
    const int    d  = (int)(gid % DI);
    const size_t bt = gid / DI;          // b*L + t
    const int    t  = (int)(bt % L_);

    const float* base = xz + bt * NXZ + d;   // xz[b,t,d] (xc half)
    float acc = cb[d];
    #pragma unroll
    for (int k = 0; k < DC; ++k) {
        const int tt = t - (DC - 1) + k;
        const float v = (tt >= 0) ? base[(ptrdiff_t)(k - (DC - 1)) * NXZ] : 0.f;
        acc += cw[d * DC + k] * v;
    }
    const float s = acc / (1.f + __expf(-acc));   // silu
    xc_f[gid] = s;
    xc_b[gid] = (__bf16)s;
}

// ============================================================
// selective scan: one wave32 per (b,d); lane = state index n (DS==32)
// ============================================================
__global__ __launch_bounds__(256) void scan_kernel(
    const float* __restrict__ xp, const float* __restrict__ xc,
    const float* __restrict__ A_log, const float* __restrict__ Dp,
    float* __restrict__ y)
{
    const int lane = threadIdx.x & 31;
    const int wid  = (int)((blockIdx.x * 256 + threadIdx.x) >> 5); // 0..B*DI-1
    const int b = wid / DI;
    const int d = wid % DI;

    const float An = -__expf(A_log[d * DS + lane]);
    const float Dd = Dp[d];

    const float* xpb = xp + (size_t)b * L_ * NP;
    const float* xcb = xc + (size_t)b * L_ * DI;
    float*       yb  = y  + (size_t)b * L_ * DI;

    float h = 0.f;
    for (int t = 0; t < L_; ++t) {
        const float* xpt = xpb + (size_t)t * NP;
        const float dl    = xpt[2 * DS + d];
        const float delta = (dl > 20.f) ? dl : log1pf(__expf(dl));   // softplus
        const float xcv   = xcb[(size_t)t * DI + d];
        const float Bn    = xpt[lane];
        const float Cn    = xpt[DS + lane];

        h = __expf(delta * An) * h + (delta * xcv) * Bn;

        float yv = h * Cn;
        #pragma unroll
        for (int off = 16; off; off >>= 1) yv += __shfl_xor(yv, off, 32);
        if (lane == 0) yb[(size_t)t * DI + d] = yv + xcv * Dd;
    }
}

// ============================================================
// y * silu(z) -> bf16 (input to GEMM3)
// ============================================================
__global__ __launch_bounds__(256) void gate_kernel(
    const float* __restrict__ y, const float* __restrict__ xz,
    __bf16* __restrict__ out)
{
    const size_t gid = (size_t)blockIdx.x * 256 + threadIdx.x;
    if (gid >= (size_t)BL * DI) return;
    const int    d  = (int)(gid % DI);
    const size_t bt = gid / DI;
    const float z = xz[bt * NXZ + DI + d];
    const float s = z / (1.f + __expf(-z));
    out[gid] = (__bf16)(y[gid] * s);
}

// ============================================================
// host-side orchestration
// ============================================================
extern "C" void kernel_launch(void* const* d_in, const int* in_sizes, int n_in,
                              void* d_out, int out_size, void* d_ws, size_t ws_size,
                              hipStream_t stream)
{
    (void)in_sizes; (void)n_in; (void)out_size; (void)ws_size;

    const float* x      = (const float*)d_in[0];
    const float* Win    = (const float*)d_in[1];
    const float* conv_w = (const float*)d_in[2];
    const float* conv_b = (const float*)d_in[3];
    const float* Wx     = (const float*)d_in[4];
    const float* A_log  = (const float*)d_in[5];
    const float* Dvec   = (const float*)d_in[6];
    const float* Wout   = (const float*)d_in[7];
    const float* ln_g   = (const float*)d_in[8];
    const float* ln_b   = (const float*)d_in[9];
    float* out = (float*)d_out;

    // ---- workspace layout ----
    char* ws = (char*)d_ws;
    size_t off = 0;
    auto alloc = [&](size_t bytes) -> void* {
        void* p = ws + off;
        off += (bytes + 255) & ~(size_t)255;
        return p;
    };
    __bf16* xn_bf   = (__bf16*)alloc((size_t)BL * DM * 2);
    __bf16* win_bf  = (__bf16*)alloc((size_t)NXZ * DM * 2);
    __bf16* wx_bf   = (__bf16*)alloc((size_t)NP * DI * 2);
    __bf16* wout_bf = (__bf16*)alloc((size_t)DM * DI * 2);
    float*  xz      = (float*) alloc((size_t)BL * NXZ * 4);
    float*  xc_f    = (float*) alloc((size_t)BL * DI * 4);
    __bf16* xc_bf   = (__bf16*)alloc((size_t)BL * DI * 2);
    float*  xp      = (float*) alloc((size_t)BL * NP * 4);
    float*  y_f     = (float*) alloc((size_t)BL * DI * 4);
    __bf16* yg_bf   = (__bf16*)alloc((size_t)BL * DI * 2);

    // 1) weight converts
    {
        int n;
        n = NXZ * DM; cvt_f32_bf16_kernel<<<(n + 255) / 256, 256, 0, stream>>>(Win,  win_bf,  n);
        n = NP  * DI; cvt_f32_bf16_kernel<<<(n + 255) / 256, 256, 0, stream>>>(Wx,   wx_bf,   n);
        n = DM  * DI; cvt_f32_bf16_kernel<<<(n + 255) / 256, 256, 0, stream>>>(Wout, wout_bf, n);
    }

    // 2) LayerNorm -> bf16
    layernorm_bf16_kernel<<<BL, 256, 0, stream>>>(x, ln_g, ln_b, xn_bf);

    // 3) GEMM1: xz = xn @ Win^T   (4096 x 3072, K=768)
    gemm_bf16_wmma_kernel<<<(BL / BM) * (NXZ / BN), 256, 0, stream>>>(
        xn_bf, win_bf, xz, nullptr, BL, NXZ, DM);

    // 4) depthwise conv + SiLU
    {
        size_t n = (size_t)BL * DI;
        conv_silu_kernel<<<(unsigned)((n + 255) / 256), 256, 0, stream>>>(
            xz, conv_w, conv_b, xc_f, xc_bf);
    }

    // 5) GEMM2: xp = xc @ Wx^T   (4096 x 1600, K=1536)
    gemm_bf16_wmma_kernel<<<(BL / BM) * (NP / BN), 256, 0, stream>>>(
        xc_bf, wx_bf, xp, nullptr, BL, NP, DI);

    // 6) selective scan: B*DI waves
    scan_kernel<<<(B_ * DI) / 8, 256, 0, stream>>>(xp, xc_f, A_log, Dvec, y_f);

    // 7) gate with silu(z)
    {
        size_t n = (size_t)BL * DI;
        gate_kernel<<<(unsigned)((n + 255) / 256), 256, 0, stream>>>(y_f, xz, yg_bf);
    }

    // 8) GEMM3: out = yg @ Wout^T + residual   (4096 x 768, K=1536)
    gemm_bf16_wmma_kernel<<<(BL / BM) * (DM / BN), 256, 0, stream>>>(
        yg_bf, wout_bf, out, x, BL, DM, DI);
}
